// Nova_Block_20779051778114
// MI455X (gfx1250) — compile-verified, matrix operand order checked by reading, not run on hardware
//
#include <hip/hip_runtime.h>
#include <hip/hip_bf16.h>
#include <math.h>

typedef __bf16 bf16_t;
typedef __attribute__((ext_vector_type(16))) __bf16 v16bf;
typedef __attribute__((ext_vector_type(8)))  __bf16 v8bf;
typedef __attribute__((ext_vector_type(8)))  float  v8f;

#define H_DIM 1024
#define B_DIM 4
#define S_DIM 1024
#define NS_DIM 2
#define NH 16
#define DH 64
#define NROWS (B_DIM * S_DIM)   // 4096

// ---------------------------------------------------------------------------
// Weight convert + transpose: wt[n*H + k] = bf16(w[k*H + n])
// ---------------------------------------------------------------------------
__global__ __launch_bounds__(256) void wcvt_kernel(const float* __restrict__ w,
                                                   bf16_t* __restrict__ wt) {
    int idx = blockIdx.x * 256 + threadIdx.x;          // 0 .. H*H-1
    int n = idx >> 10;                                 // / H
    int k = idx & (H_DIM - 1);                         // % H
    wt[(size_t)n * H_DIM + k] = (bf16_t)w[(size_t)k * H_DIM + n];
}

// ---------------------------------------------------------------------------
// Block reduction helper (blockDim.x == 256)
// ---------------------------------------------------------------------------
static __device__ __forceinline__ float block_reduce_sum(float v, float* red) {
    int tid = threadIdx.x;
    red[tid] = v;
    __syncthreads();
    for (int s = 128; s > 0; s >>= 1) {
        if (tid < s) red[tid] += red[tid + s];
        __syncthreads();
    }
    float r = red[0];
    __syncthreads();
    return r;
}

// ---------------------------------------------------------------------------
// Fused input LN + mean fusion over [id, side0, side1]
// Writes x = LN(id) (bf16) and fused = (LN(id)+LN(s0)+LN(s1))/3 (bf16)
// ---------------------------------------------------------------------------
__global__ __launch_bounds__(256) void fuse_ln_kernel(
    const float* __restrict__ id, const float* __restrict__ side,
    const float* __restrict__ g_id, const float* __restrict__ b_id,
    const float* __restrict__ g_s,  const float* __restrict__ b_s,
    bf16_t* __restrict__ x_bf, bf16_t* __restrict__ fused_bf) {
    __shared__ float red[256];
    const int row = blockIdx.x;
    const int tid = threadIdx.x;
    const float invH = 1.0f / (float)H_DIM;

    float acc[4];
    {
        const float* sp = id + (size_t)row * H_DIM;
        float v[4], s = 0.f, sq = 0.f;
        #pragma unroll
        for (int i = 0; i < 4; ++i) {
            int c = tid + i * 256;
            v[i] = sp[c]; s += v[i]; sq += v[i] * v[i];
        }
        s  = block_reduce_sum(s,  red);
        sq = block_reduce_sum(sq, red);
        float mean = s * invH;
        float var  = sq * invH - mean * mean;
        float rstd = rsqrtf(var + 1e-5f);
        #pragma unroll
        for (int i = 0; i < 4; ++i) {
            int c = tid + i * 256;
            float xn = (v[i] - mean) * rstd * g_id[c] + b_id[c];
            x_bf[(size_t)row * H_DIM + c] = (bf16_t)xn;
            acc[i] = xn;
        }
    }
    for (int ns = 0; ns < NS_DIM; ++ns) {
        const float* sp = side + ((size_t)row * NS_DIM + ns) * H_DIM;
        const float* g = g_s + ns * H_DIM;
        const float* b = b_s + ns * H_DIM;
        float v[4], s = 0.f, sq = 0.f;
        #pragma unroll
        for (int i = 0; i < 4; ++i) {
            int c = tid + i * 256;
            v[i] = sp[c]; s += v[i]; sq += v[i] * v[i];
        }
        s  = block_reduce_sum(s,  red);
        sq = block_reduce_sum(sq, red);
        float mean = s * invH;
        float var  = sq * invH - mean * mean;
        float rstd = rsqrtf(var + 1e-5f);
        #pragma unroll
        for (int i = 0; i < 4; ++i) {
            int c = tid + i * 256;
            acc[i] += (v[i] - mean) * rstd * g[c] + b[c];
        }
    }
    const float invF = 1.0f / (float)(NS_DIM + 1);
    #pragma unroll
    for (int i = 0; i < 4; ++i) {
        int c = tid + i * 256;
        fused_bf[(size_t)row * H_DIM + c] = (bf16_t)(acc[i] * invF);
    }
}

// ---------------------------------------------------------------------------
// Row LayerNorm (f32 in -> bf16 out), runtime eps (ff_ln uses 1e-6)
// ---------------------------------------------------------------------------
__global__ __launch_bounds__(256) void row_ln_kernel(
    const float* __restrict__ src, const float* __restrict__ g,
    const float* __restrict__ b, float eps, bf16_t* __restrict__ dst) {
    __shared__ float red[256];
    const int row = blockIdx.x;
    const int tid = threadIdx.x;
    const float invH = 1.0f / (float)H_DIM;
    const float* sp = src + (size_t)row * H_DIM;
    float v[4], s = 0.f, sq = 0.f;
    #pragma unroll
    for (int i = 0; i < 4; ++i) {
        int c = tid + i * 256;
        v[i] = sp[c]; s += v[i]; sq += v[i] * v[i];
    }
    s  = block_reduce_sum(s,  red);
    sq = block_reduce_sum(sq, red);
    float mean = s * invH;
    float var  = sq * invH - mean * mean;
    float rstd = rsqrtf(var + eps);
    #pragma unroll
    for (int i = 0; i < 4; ++i) {
        int c = tid + i * 256;
        dst[(size_t)row * H_DIM + c] = (bf16_t)((v[i] - mean) * rstd * g[c] + b[c]);
    }
}

// ---------------------------------------------------------------------------
// Tiled WMMA bf16 GEMM:  out[N x M] = A[N x K] * B[K x M]  (B given as B^T, M x K)
// 128 threads = 4 waves; wave w computes a 32x64 tile:
//   rows [blk.x*128 + w*32, +32), cols [blk.y*64, +64)
// 8 WMMAs per K-step against 6 fragment loads (A reused x4, B reused x2).
// MODE 0: store bf16 row-major
// MODE 1: out_f32 = acc + bias[col] + resid[row,col]
// MODE 2: store bf16( gelu_erf(acc + bias[col]) )
// MODE 3: store bf16 TRANSPOSED: out[col * NROWS + row] (for V^T), packed 8-wide
// ---------------------------------------------------------------------------
template <int MODE>
__global__ __launch_bounds__(128) void gemm_bf16_kernel(
    const bf16_t* __restrict__ A, const bf16_t* __restrict__ BT,
    const float* __restrict__ bias, const float* __restrict__ resid,
    void* __restrict__ outp, int Kdim, int Mcols) {
    const int lane  = threadIdx.x & 31;
    const int wave  = threadIdx.x >> 5;
    const int row0  = blockIdx.x * 128 + wave * 32;
    const int col0  = blockIdx.y * 64;
    const int lrow  = lane & 15;
    const int khalf = (lane >> 4) * 16;   // lanes 16-31 carry K+16..K+31

    v8f acc[2][4] = {{v8f{}, v8f{}, v8f{}, v8f{}}, {v8f{}, v8f{}, v8f{}, v8f{}}};
    const bf16_t* ap0 = A + (size_t)(row0 + lrow) * Kdim + khalf;
    const bf16_t* ap1 = A + (size_t)(row0 + 16 + lrow) * Kdim + khalf;

    for (int k0 = 0; k0 < Kdim; k0 += 32) {
        v16bf a0 = *(const v16bf*)(ap0 + k0);
        v16bf a1 = *(const v16bf*)(ap1 + k0);
        if (k0 + 256 < Kdim) {  // gfx1250 global_prefetch_b8 a few steps ahead
            __builtin_prefetch(ap0 + k0 + 256, 0, 1);
        }
        #pragma unroll
        for (int t = 0; t < 4; ++t) {
            const bf16_t* bp =
                BT + (size_t)(col0 + t * 16 + lrow) * Kdim + khalf + k0;
            v16bf b = *(const v16bf*)bp;
            acc[0][t] = __builtin_amdgcn_wmma_f32_16x16x32_bf16(
                false, a0, false, b, (short)0, acc[0][t], false, false);
            acc[1][t] = __builtin_amdgcn_wmma_f32_16x16x32_bf16(
                false, a1, false, b, (short)0, acc[1][t], false, false);
        }
    }

    const int mbase = (lane >> 4) * 8;
    #pragma unroll
    for (int u = 0; u < 2; ++u) {
        #pragma unroll
        for (int t = 0; t < 4; ++t) {
            const int col   = col0 + t * 16 + lrow;
            const int rbase = row0 + u * 16 + mbase;
            if (MODE == 3) {
                // V^T: 8 consecutive rows -> one packed 16B store per lane
                v8bf pk;
                #pragma unroll
                for (int r = 0; r < 8; ++r) pk[r] = (bf16_t)acc[u][t][r];
                *(v8bf*)((bf16_t*)outp + (size_t)col * NROWS + rbase) = pk;
            } else {
                #pragma unroll
                for (int r = 0; r < 8; ++r) {
                    int row = rbase + r;
                    float vv = acc[u][t][r];
                    if (MODE == 0) {
                        ((bf16_t*)outp)[(size_t)row * Mcols + col] = (bf16_t)vv;
                    } else if (MODE == 1) {
                        vv += bias[col] + resid[(size_t)row * Mcols + col];
                        ((float*)outp)[(size_t)row * Mcols + col] = vv;
                    } else {  // MODE 2: bias + exact-erf GELU
                        vv += bias[col];
                        vv = 0.5f * vv * (1.0f + erff(vv * 0.70710678118654752f));
                        ((bf16_t*)outp)[(size_t)row * Mcols + col] = (bf16_t)vv;
                    }
                }
            }
        }
    }
}

// ---------------------------------------------------------------------------
// Attention: one block per (b, h, 16-query tile). 4 waves, 64KB LDS score strip.
//   scores = Q K^T / sqrt(H)  -> softmax over 1024 keys -> O = P V
// q/k are bf16 [NROWS x H]; vt is bf16 V^T: vt[(h*64+d) * NROWS + (b*S + key)].
// ---------------------------------------------------------------------------
__global__ __launch_bounds__(128) void attn_kernel(
    const bf16_t* __restrict__ q, const bf16_t* __restrict__ k,
    const bf16_t* __restrict__ vt, bf16_t* __restrict__ o) {
    __shared__ float sS[16 * S_DIM];  // 64 KB score strip (f32)

    const int bid  = blockIdx.x;
    const int qt   = bid & 63;
    const int h    = (bid >> 6) & (NH - 1);
    const int bb   = bid >> 10;
    const int lane = threadIdx.x & 31;
    const int wave = threadIdx.x >> 5;
    const int lrow = lane & 15;
    const int khalf = (lane >> 4) * 16;
    const size_t brow = (size_t)bb * S_DIM;
    const int q0 = qt * 16;
    const int hc = h * DH;

    // ---- Phase 1: scores (each wave does 16 of the 64 key tiles) ----
    const bf16_t* qp = q + (brow + q0 + lrow) * H_DIM + hc + khalf;
    v16bf a0 = *(const v16bf*)(qp);
    v16bf a1 = *(const v16bf*)(qp + 32);
    for (int kt = wave * 16; kt < wave * 16 + 16; ++kt) {
        v8f acc = {};
        const bf16_t* kp = k + (brow + kt * 16 + lrow) * H_DIM + hc + khalf;
        v16bf b0 = *(const v16bf*)(kp);
        v16bf b1 = *(const v16bf*)(kp + 32);
        acc = __builtin_amdgcn_wmma_f32_16x16x32_bf16(false, a0, false, b0,
                                                      (short)0, acc, false, false);
        acc = __builtin_amdgcn_wmma_f32_16x16x32_bf16(false, a1, false, b1,
                                                      (short)0, acc, false, false);
        #pragma unroll
        for (int r = 0; r < 8; ++r) {
            int m = (lane >> 4) * 8 + r;
            sS[m * S_DIM + kt * 16 + lrow] = acc[r] * 0.03125f;  // 1/sqrt(1024)
        }
    }
    __syncthreads();

    // ---- Phase 2: softmax in place (8 lanes per row, shuffle reductions) ----
    {
        const int tid = threadIdx.x;
        const int row = tid >> 3;   // 0..15
        const int seg = tid & 7;    // 8 consecutive lanes share a row
        float* p = sS + row * S_DIM + seg * 128;
        float mx = -3.4e38f;
        for (int c = 0; c < 128; ++c) mx = fmaxf(mx, p[c]);
        for (int d = 1; d < 8; d <<= 1) mx = fmaxf(mx, __shfl_xor(mx, d, 32));
        float sum = 0.f;
        for (int c = 0; c < 128; ++c) {
            float e = expf(p[c] - mx);
            p[c] = e;
            sum += e;
        }
        for (int d = 1; d < 8; d <<= 1) sum += __shfl_xor(sum, d, 32);
        float inv = 1.0f / sum;
        for (int c = 0; c < 128; ++c) p[c] *= inv;
    }
    __syncthreads();

    // ---- Phase 3: O = P V (each wave owns a 16-wide slice of dh) ----
    {
        v8f acc = {};
        const int dcol = hc + wave * 16 + lrow;
        const bf16_t* vrow = vt + (size_t)dcol * NROWS + brow + khalf;
        const float4* prow4 =
            (const float4*)(sS + lrow * S_DIM + khalf);
        for (int kt = 0; kt < 32; ++kt) {          // 32-key steps
            v16bf a;
            #pragma unroll
            for (int jj = 0; jj < 4; ++jj) {       // ds_load_b128 reads of P
                float4 f = prow4[kt * 8 + jj];
                a[4 * jj + 0] = (bf16_t)f.x;
                a[4 * jj + 1] = (bf16_t)f.y;
                a[4 * jj + 2] = (bf16_t)f.z;
                a[4 * jj + 3] = (bf16_t)f.w;
            }
            v16bf b = *(const v16bf*)(vrow + kt * 32);  // contiguous V^T load
            acc = __builtin_amdgcn_wmma_f32_16x16x32_bf16(
                false, a, false, b, (short)0, acc, false, false);
        }
        #pragma unroll
        for (int r = 0; r < 8; ++r) {
            int m = (lane >> 4) * 8 + r;
            o[(brow + q0 + m) * H_DIM + hc + wave * 16 + lrow] = (bf16_t)acc[r];
        }
    }
}

// ---------------------------------------------------------------------------
// Launcher
// ---------------------------------------------------------------------------
extern "C" void kernel_launch(void* const* d_in, const int* in_sizes, int n_in,
                              void* d_out, int out_size, void* d_ws, size_t ws_size,
                              hipStream_t stream) {
    const float* id      = (const float*)d_in[0];
    const float* side    = (const float*)d_in[1];
    const float* id_ln_g = (const float*)d_in[2];
    const float* id_ln_b = (const float*)d_in[3];
    const float* s_ln_g  = (const float*)d_in[4];
    const float* s_ln_b  = (const float*)d_in[5];
    const float* wq      = (const float*)d_in[6];
    const float* wk      = (const float*)d_in[7];
    const float* wv      = (const float*)d_in[8];
    const float* wo      = (const float*)d_in[9];
    const float* bo      = (const float*)d_in[10];
    const float* ff_ln_g = (const float*)d_in[11];
    const float* ff_ln_b = (const float*)d_in[12];
    const float* w1      = (const float*)d_in[13];
    const float* b1      = (const float*)d_in[14];
    const float* w2      = (const float*)d_in[15];
    const float* b2      = (const float*)d_in[16];
    float* out           = (float*)d_out;

    // Workspace layout (bytes). Requires ~92 MB scratch.
    char* ws = (char*)d_ws;
    const size_t WSZ = (size_t)H_DIM * H_DIM * sizeof(bf16_t);   // 2 MB per weight
    const size_t ASZ = (size_t)NROWS * H_DIM * sizeof(bf16_t);   // 8 MB per bf16 activation
    bf16_t* wq_t   = (bf16_t*)(ws + 0 * WSZ);
    bf16_t* wk_t   = (bf16_t*)(ws + 1 * WSZ);
    bf16_t* wv_t   = (bf16_t*)(ws + 2 * WSZ);
    bf16_t* wo_t   = (bf16_t*)(ws + 3 * WSZ);
    bf16_t* w1_t   = (bf16_t*)(ws + 4 * WSZ);
    bf16_t* w2_t   = (bf16_t*)(ws + 5 * WSZ);
    char* act      = ws + 6 * WSZ;
    bf16_t* x_bf   = (bf16_t*)(act + 0 * ASZ);
    bf16_t* f_bf   = (bf16_t*)(act + 1 * ASZ);
    bf16_t* q_bf   = (bf16_t*)(act + 2 * ASZ);
    bf16_t* k_bf   = (bf16_t*)(act + 3 * ASZ);
    bf16_t* vt_bf  = (bf16_t*)(act + 4 * ASZ);   // V^T: [H_DIM x NROWS]
    bf16_t* o_bf   = (bf16_t*)(act + 5 * ASZ);
    float*  out_f  = (float*)(act + 6 * ASZ);                    // 16 MB
    bf16_t* ffi_bf = (bf16_t*)(act + 6 * ASZ + (size_t)NROWS * H_DIM * 4);
    bf16_t* y1_bf  = (bf16_t*)((char*)ffi_bf + ASZ);

    const int WBLK = (H_DIM * H_DIM) / 256;  // 4096
    wcvt_kernel<<<WBLK, 256, 0, stream>>>(wq, wq_t);
    wcvt_kernel<<<WBLK, 256, 0, stream>>>(wk, wk_t);
    wcvt_kernel<<<WBLK, 256, 0, stream>>>(wv, wv_t);
    wcvt_kernel<<<WBLK, 256, 0, stream>>>(wo, wo_t);
    wcvt_kernel<<<WBLK, 256, 0, stream>>>(w1, w1_t);
    wcvt_kernel<<<WBLK, 256, 0, stream>>>(w2, w2_t);

    fuse_ln_kernel<<<NROWS, 256, 0, stream>>>(id, side, id_ln_g, id_ln_b,
                                              s_ln_g, s_ln_b, x_bf, f_bf);

    dim3 gg(NROWS / 128, H_DIM / 64);  // (32, 16)
    // Q = fused @ wq ; K = fused @ wk ; V^T = (x @ wv)^T
    gemm_bf16_kernel<0><<<gg, 128, 0, stream>>>(f_bf, wq_t, nullptr, nullptr,
                                                q_bf, H_DIM, H_DIM);
    gemm_bf16_kernel<0><<<gg, 128, 0, stream>>>(f_bf, wk_t, nullptr, nullptr,
                                                k_bf, H_DIM, H_DIM);
    gemm_bf16_kernel<3><<<gg, 128, 0, stream>>>(x_bf, wv_t, nullptr, nullptr,
                                                vt_bf, H_DIM, H_DIM);

    // Attention: B * NH * (S/16) blocks
    attn_kernel<<<B_DIM * NH * (S_DIM / 16), 128, 0, stream>>>(q_bf, k_bf, vt_bf, o_bf);

    // out = id + o @ wo + bo   (f32)
    gemm_bf16_kernel<1><<<gg, 128, 0, stream>>>(o_bf, wo_t, bo, id, out_f,
                                                H_DIM, H_DIM);

    // FF: y = LN(out) ; y = gelu(y@w1+b1) ; final = out + y@w2 + b2
    row_ln_kernel<<<NROWS, 256, 0, stream>>>(out_f, ff_ln_g, ff_ln_b, 1e-6f, ffi_bf);
    gemm_bf16_kernel<2><<<gg, 128, 0, stream>>>(ffi_bf, w1_t, b1, nullptr,
                                                y1_bf, H_DIM, H_DIM);
    gemm_bf16_kernel<1><<<gg, 128, 0, stream>>>(y1_bf, w2_t, b2, out_f, out,
                                                H_DIM, H_DIM);
    (void)in_sizes; (void)n_in; (void)out_size; (void)ws_size;
}
